// SingleLSTMModel_51668456571405
// MI455X (gfx1250) — compile-verified, matrix operand order checked by reading, not run on hardware
//
#include <hip/hip_runtime.h>
#include <math.h>

// ---------------- types for CDNA5 WMMA ----------------
typedef __attribute__((ext_vector_type(16))) __bf16 v16bf;
typedef __attribute__((ext_vector_type(8)))  float  v8f;

// Problem constants (from reference)
#define BB 256
#define TT 512
#define EE 128
#define HH 512
#define AA 10
#define KK (HH + 2 * EE)   // 768
#define NBLOCKS 64         // persistent grid: 16 m-tiles x 4 n-groups

static __device__ __forceinline__ unsigned short f32_to_bf16(float f) {
    union { float f; unsigned u; } v; v.f = f;
    unsigned r = v.u + 0x7FFFu + ((v.u >> 16) & 1u);   // round-to-nearest-even
    return (unsigned short)(r >> 16);
}

static __device__ __forceinline__ float sigmoidf_(float x) {
    return 1.0f / (1.0f + __expf(-x));
}

static __device__ __forceinline__ float wave_sum(float v) {
    #pragma unroll
    for (int off = 16; off > 0; off >>= 1) v += __shfl_xor(v, off, 32);
    return v;
}

// Device-wide generation barrier (counter monotonically increases; no reset
// race). counter/gen live in d_ws and are zeroed by prep_kernel every call.
static __device__ __forceinline__ void grid_sync(unsigned* counter, unsigned* gen,
                                                 unsigned it) {
    __syncthreads();
    if (threadIdx.x == 0) {
        __threadfence();   // release all prior global writes device-wide
        unsigned prev = __hip_atomic_fetch_add(counter, 1u, __ATOMIC_ACQ_REL,
                                               __HIP_MEMORY_SCOPE_AGENT);
        if (prev == (unsigned)NBLOCKS * (it + 1u) - 1u) {
            __hip_atomic_store(gen, it + 1u, __ATOMIC_RELEASE,
                               __HIP_MEMORY_SCOPE_AGENT);
        } else {
            while (__hip_atomic_load(gen, __ATOMIC_ACQUIRE,
                                     __HIP_MEMORY_SCOPE_AGENT) < it + 1u) {
                __builtin_amdgcn_s_sleep(2);
            }
        }
    }
    __syncthreads();
}

// ---------------------------------------------------------------------------
// prep: bf16-transpose gate weights into [gate*512 + n][k] (k contiguous),
//       bf16-transpose W_lin into [n][k], zero h (both buffers), c, barrier.
// ---------------------------------------------------------------------------
__global__ void prep_kernel(const float* __restrict__ Wf, const float* __restrict__ Wi,
                            const float* __restrict__ Wc, const float* __restrict__ Wo,
                            const float* __restrict__ Wlin,
                            unsigned short* __restrict__ WtBig,
                            unsigned short* __restrict__ WtLin,
                            float* __restrict__ h0, float* __restrict__ h1,
                            float* __restrict__ c, unsigned* __restrict__ sync) {
    int idx = blockIdx.x * blockDim.x + threadIdx.x;
    int stride = gridDim.x * blockDim.x;

    for (int i = idx; i < 4 * HH * KK; i += stride) {
        int ng = i / KK;          // 0..2047
        int k  = i - ng * KK;     // 0..767
        int g  = ng >> 9;         // gate
        int n  = ng & (HH - 1);   // column within gate
        float w;
        if      (g == 0) w = Wf[k * HH + n];
        else if (g == 1) w = Wi[k * HH + n];
        else if (g == 2) w = Wc[k * HH + n];
        else             w = Wo[k * HH + n];
        WtBig[i] = f32_to_bf16(w);
    }
    for (int i = idx; i < EE * HH; i += stride) {
        int n = i / HH;
        int k = i - n * HH;
        WtLin[i] = f32_to_bf16(Wlin[k * EE + n]);
    }
    for (int i = idx; i < BB * HH; i += stride) { h0[i] = 0.0f; h1[i] = 0.0f; c[i] = 0.0f; }
    if (idx < 2) sync[idx] = 0u;
}

// ---------------------------------------------------------------------------
// Persistent recurrence kernel: 64 blocks x 256 threads stay resident for all
// T=512 steps. Block (mtile = blk&15, ngrp = blk>>4). Per step:
//   Phase A: compute attention for my 16 batch rows (wave-parallel, shfl
//            reductions) and build the 16x768 bf16 z-tile in LDS.
//   Phase B: each of 8 waves does the 4-gate 16x16xK GEMM for one n-tile
//            (A from LDS via ds_load, B streamed from L2), then the fused
//            sigmoid/tanh LSTM state update + masked bf16 h_t store.
//   grid_sync once per step (h is double-buffered: read h[t&1], write h[~t&1]).
// ---------------------------------------------------------------------------
__global__ __launch_bounds__(256)
void lstm_persistent(const float* __restrict__ x1, const float* __restrict__ x2,
                     const int* __restrict__ seq,
                     const unsigned short* __restrict__ Wt,
                     const float* __restrict__ bfv, const float* __restrict__ biv,
                     const float* __restrict__ bcv, const float* __restrict__ bov,
                     const float* __restrict__ Wattn,
                     float* __restrict__ h0buf, float* __restrict__ h1buf,
                     float* __restrict__ c,
                     unsigned short* __restrict__ hs,
                     unsigned* __restrict__ counter, unsigned* __restrict__ gen) {
    __shared__ __align__(32) unsigned short zt[16 * KK];   // 24 KB z tile

    const int blk   = blockIdx.x;
    const int mtile = blk & 15;
    const int ngrp  = blk >> 4;          // 0..3 -> n-tiles ngrp*8 .. +7
    const int wv    = threadIdx.x >> 5;  // 0..7
    const int lane  = threadIdx.x & 31;

    // --- fixed per-wave GEMM geometry ---
    const int ntile = ngrp * 8 + wv;                 // 0..31
    const int ncol  = ntile * 16 + (lane & 15);      // hidden unit
    const int ksel  = (lane >> 4) * 16;              // K half select
    const unsigned short* bF = Wt + ((size_t)(0 * HH + ncol)) * KK + ksel;
    const unsigned short* bI = Wt + ((size_t)(1 * HH + ncol)) * KK + ksel;
    const unsigned short* bC = Wt + ((size_t)(2 * HH + ncol)) * KK + ksel;
    const unsigned short* bO = Wt + ((size_t)(3 * HH + ncol)) * KK + ksel;
    const unsigned short* aL = zt + (size_t)(lane & 15) * KK + ksel;
    const float bF0 = bfv[ncol], bI0 = biv[ncol], bC0 = bcv[ncol], bO0 = bov[ncol];
    const int rbase = (lane >> 4) * 8;

    for (int t = 0; t < TT; ++t) {
        const float* hin  = (t & 1) ? h1buf : h0buf;
        float*       hout = (t & 1) ? h0buf : h1buf;

        // ---------------- Phase A: attention + z tile ----------------
        #pragma unroll
        for (int rr = 0; rr < 2; ++rr) {
            const int r = wv * 2 + rr;               // row in tile, 0..15
            const int b = mtile * 16 + r;            // batch index
            const float* hb  = hin + (size_t)b * HH;
            const float* x1t = x1 + ((size_t)b * TT + t) * EE;
            const float* x2t = x2 + (((size_t)b * TT + t) * AA) * EE;

            // base score: dot(concat(h, x1_t), w_attn[:640])
            float p = 0.0f;
            for (int k = lane; k < HH + EE; k += 32)
                p += ((k < HH) ? hb[k] : x1t[k - HH]) * Wattn[k];
            p = wave_sum(p);

            // slot scores (all lanes hold the reduced value)
            float attnv[AA];
            float mx = -3.4e38f;
            #pragma unroll
            for (int a = 0; a < AA; ++a) {
                float q = 0.0f;
                for (int e = lane; e < EE; e += 32)
                    q += x2t[a * EE + e] * Wattn[HH + EE + e];
                q = wave_sum(q) + p;
                attnv[a] = q;
                mx = fmaxf(mx, q);
            }
            float s = 0.0f;
            #pragma unroll
            for (int a = 0; a < AA; ++a) { attnv[a] = __expf(attnv[a] - mx); s += attnv[a]; }
            const float inv = 1.0f / s;

            // z row = bf16([h, x1_t, ctx]) into LDS
            unsigned short* zr = zt + (size_t)r * KK;
            for (int k = lane; k < HH; k += 32) zr[k] = f32_to_bf16(hb[k]);
            for (int e = lane; e < EE; e += 32) {
                zr[HH + e] = f32_to_bf16(x1t[e]);
                float ctx = 0.0f;
                #pragma unroll
                for (int a = 0; a < AA; ++a) ctx += attnv[a] * x2t[a * EE + e];
                zr[HH + EE + e] = f32_to_bf16(ctx * inv);
            }
        }
        __syncthreads();

        // ---------------- Phase B: 4-gate WMMA GEMM ----------------
        v8f accF = {}, accI = {}, accC = {}, accO = {};
        v16bf a  = *reinterpret_cast<const v16bf*>(aL);
        v16bf wf = *reinterpret_cast<const v16bf*>(bF);
        v16bf wi = *reinterpret_cast<const v16bf*>(bI);
        v16bf wc = *reinterpret_cast<const v16bf*>(bC);
        v16bf wo = *reinterpret_cast<const v16bf*>(bO);
        #pragma unroll 4
        for (int ks = 0; ks < KK / 32; ++ks) {
            const int kn = (ks + 1) * 32;
            v16bf a2 = a, wf2 = wf, wi2 = wi, wc2 = wc, wo2 = wo;
            if (kn < KK) {   // software pipeline: prefetch next K-step
                a2  = *reinterpret_cast<const v16bf*>(aL + kn);
                wf2 = *reinterpret_cast<const v16bf*>(bF + kn);
                wi2 = *reinterpret_cast<const v16bf*>(bI + kn);
                wc2 = *reinterpret_cast<const v16bf*>(bC + kn);
                wo2 = *reinterpret_cast<const v16bf*>(bO + kn);
            }
            accF = __builtin_amdgcn_wmma_f32_16x16x32_bf16(false, a, false, wf, (short)0, accF, false, false);
            accI = __builtin_amdgcn_wmma_f32_16x16x32_bf16(false, a, false, wi, (short)0, accI, false, false);
            accC = __builtin_amdgcn_wmma_f32_16x16x32_bf16(false, a, false, wc, (short)0, accC, false, false);
            accO = __builtin_amdgcn_wmma_f32_16x16x32_bf16(false, a, false, wo, (short)0, accO, false, false);
            a = a2; wf = wf2; wi = wi2; wc = wc2; wo = wo2;
        }

        // fused elementwise LSTM update on the 16x16 tile
        #pragma unroll
        for (int r = 0; r < 8; ++r) {
            const int m   = mtile * 16 + rbase + r;   // batch index
            const int idx = m * HH + ncol;
            float fg = sigmoidf_(accF[r] + bF0);
            float ig = sigmoidf_(accI[r] + bI0);
            float gg = tanhf(accC[r] + bC0);
            float og = sigmoidf_(accO[r] + bO0);
            float cold = c[idx];
            float cnew = fg * cold + ig * gg;
            float hnew = og * tanhf(cnew);
            bool  msk  = t < seq[m];
            float hold = hin[idx];
            hout[idx] = msk ? hnew : hold;
            c[idx]    = msk ? cnew : cold;
            hs[((size_t)m * TT + t) * HH + ncol] = f32_to_bf16(msk ? hnew : 0.0f);
        }

        grid_sync(counter, gen, (unsigned)t);
    }
}

// ---------------------------------------------------------------------------
// proj: outputs = hs(B*T, 512) @ W_lin(512,128) + b_lin -> (B*T, 128) f32
// Each wave: one 16x16 tile, K=512 (16 WMMA). 65536 waves = 8192 blocks.
// ---------------------------------------------------------------------------
__global__ __launch_bounds__(256)
void proj_kernel(const unsigned short* __restrict__ hs,
                 const unsigned short* __restrict__ WtLin,
                 const float* __restrict__ blin,
                 float* __restrict__ out) {
    const int wave  = (blockIdx.x * blockDim.x + threadIdx.x) >> 5;
    const int lane  = threadIdx.x & 31;
    const int ntile = wave & 7;     // 8 n-tiles (N=128)
    const int mtile = wave >> 3;    // 8192 m-tiles (M=131072)

    const int arow = mtile * 16 + (lane & 15);
    const int ksel = (lane >> 4) * 16;
    const unsigned short* ap = hs + (size_t)arow * HH + ksel;

    const int ncol = ntile * 16 + (lane & 15);
    const unsigned short* bp = WtLin + (size_t)ncol * HH + ksel;

    v8f acc = {};
    v16bf a = *reinterpret_cast<const v16bf*>(ap);
    v16bf b = *reinterpret_cast<const v16bf*>(bp);
    #pragma unroll 4
    for (int ks = 0; ks < HH / 32; ++ks) {
        const int kn = (ks + 1) * 32;
        v16bf a2 = a, b2 = b;
        if (kn < HH) {
            a2 = *reinterpret_cast<const v16bf*>(ap + kn);
            b2 = *reinterpret_cast<const v16bf*>(bp + kn);
        }
        acc = __builtin_amdgcn_wmma_f32_16x16x32_bf16(false, a, false, b, (short)0, acc, false, false);
        a = a2; b = b2;
    }

    const float bl = blin[ncol];
    const int rbase = (lane >> 4) * 8;
    #pragma unroll
    for (int r = 0; r < 8; ++r) {
        const int m = mtile * 16 + rbase + r;
        out[(size_t)m * EE + ncol] = acc[r] + bl;
    }
}

// ---------------------------------------------------------------------------
extern "C" void kernel_launch(void* const* d_in, const int* in_sizes, int n_in,
                              void* d_out, int out_size, void* d_ws, size_t ws_size,
                              hipStream_t stream) {
    (void)in_sizes; (void)n_in; (void)out_size; (void)ws_size;

    const float* x1    = (const float*)d_in[0];
    const float* x2    = (const float*)d_in[1];
    const int*   seq   = (const int*)d_in[2];
    const float* Wf    = (const float*)d_in[3];
    const float* bfp   = (const float*)d_in[4];
    const float* Wi    = (const float*)d_in[5];
    const float* bip   = (const float*)d_in[6];
    const float* Wc    = (const float*)d_in[7];
    const float* bcp   = (const float*)d_in[8];
    const float* Wo    = (const float*)d_in[9];
    const float* bop   = (const float*)d_in[10];
    const float* Wattn = (const float*)d_in[11];
    const float* Wlin  = (const float*)d_in[12];
    const float* blin  = (const float*)d_in[13];
    float* out = (float*)d_out;

    // workspace carve-up (all chunk sizes are 256B multiples)
    char* ws = (char*)d_ws;
    unsigned short* WtBig = (unsigned short*)ws; ws += (size_t)4 * HH * KK * 2;  // 3.0 MB
    unsigned short* WtLin = (unsigned short*)ws; ws += (size_t)EE * HH * 2;      // 128 KB
    float* h0 = (float*)ws;                      ws += (size_t)BB * HH * 4;      // 512 KB
    float* h1 = (float*)ws;                      ws += (size_t)BB * HH * 4;      // 512 KB
    float* c  = (float*)ws;                      ws += (size_t)BB * HH * 4;      // 512 KB
    unsigned* sync = (unsigned*)ws;              ws += 256;                       // barrier state
    unsigned short* hs = (unsigned short*)ws;    // B*T*H bf16 = 128 MB

    prep_kernel<<<512, 256, 0, stream>>>(Wf, Wi, Wc, Wo, Wlin, WtBig, WtLin,
                                         h0, h1, c, sync);

    lstm_persistent<<<NBLOCKS, 256, 0, stream>>>(x1, x2, seq, WtBig,
                                                 bfp, bip, bcp, bop, Wattn,
                                                 h0, h1, c, hs,
                                                 sync, sync + 1);

    proj_kernel<<<8192, 256, 0, stream>>>(hs, WtLin, blin, out);
}
